// FadeFormerCut_49744311222997
// MI455X (gfx1250) — matrix operand order repaired and checked by reading, not compiled, not run on hardware
//
#include <hip/hip_runtime.h>
#include <hip/hip_bf16.h>

typedef __attribute__((ext_vector_type(16))) __bf16 v16bf;
typedef __attribute__((ext_vector_type(8)))  __bf16 v8bf;
typedef __attribute__((ext_vector_type(8)))  float  v8f;

#define NLAYER 6
#define NH     12
#define CDIM   768
#define DHD    64
#define FDIM   3072
#define NB     8
#define NT     1024
#define NV     50257

__device__ __forceinline__ v16bf cat16(v8bf lo, v8bf hi) {
  return __builtin_shufflevector(lo, hi, 0,1,2,3,4,5,6,7,8,9,10,11,12,13,14,15);
}
__device__ __forceinline__ v8f wmma_bf16(v16bf a, v16bf b, v8f c) {
  return __builtin_amdgcn_wmma_f32_16x16x32_bf16(false, a, false, b, (short)0, c, false, false);
}

// ---------------------------------------------------------------------------
// Weight repack: f32 -> bf16, transposed to N-major so WMMA B-fragments are
// contiguous 16B loads per lane.
// ---------------------------------------------------------------------------
// dst[(l*H*D + h*D + d)*C + c] = src[((l*H + h)*C + c)*D + d]
__global__ void pack_headed(__bf16* __restrict__ dst, const float* __restrict__ src,
                            long total, int Hh, int C, int D) {
  long stride = (long)gridDim.x * blockDim.x;
  for (long i = (long)blockIdx.x * blockDim.x + threadIdx.x; i < total; i += stride) {
    int  c  = (int)(i % C);
    long ln = i / C;
    int  n  = (int)(ln % (Hh * D));
    long l  = ln / (Hh * D);
    int  hh = n / D, d = n % D;
    dst[i] = (__bf16)src[((l * Hh + hh) * C + c) * (long)D + d];
  }
}
// dst[(l*N + n)*K + k] = src[(l*K + k)*N + n]
__global__ void pack_T(__bf16* __restrict__ dst, const float* __restrict__ src,
                       long total, int K, int N) {
  long stride = (long)gridDim.x * blockDim.x;
  for (long i = (long)blockIdx.x * blockDim.x + threadIdx.x; i < total; i += stride) {
    int  k  = (int)(i % K);
    long ln = i / K;
    int  n  = (int)(ln % N);
    long l  = ln / N;
    dst[i] = (__bf16)src[(l * K + k) * (long)N + n];
  }
}

// ---------------------------------------------------------------------------
// Embedding: h[b,t,:] = tok[x[b,t],:] + pos[t,:]
// ---------------------------------------------------------------------------
__global__ void embed_kernel(const int* __restrict__ x, const float* __restrict__ tok,
                             const float* __restrict__ pos, float* __restrict__ h,
                             int T, int C) {
  int bt = blockIdx.x;
  int t  = bt % T;
  int id = x[bt];
  for (int c = threadIdx.x; c < C; c += blockDim.x)
    h[(size_t)bt * C + c] = tok[(size_t)id * C + c] + pos[(size_t)t * C + c];
}

// ---------------------------------------------------------------------------
// LayerNorm (one block per row). Writes bf16 (GEMM input) and/or f32.
// ---------------------------------------------------------------------------
__global__ __launch_bounds__(256) void ln_kernel(const float* __restrict__ in,
                                                 const float* __restrict__ w,
                                                 const float* __restrict__ bia,
                                                 __bf16* __restrict__ outb,
                                                 float* __restrict__ outf, int C) {
  long row = blockIdx.x;
  const float* px = in + row * C;
  float s = 0.f, s2 = 0.f;
  for (int c = threadIdx.x; c < C; c += 256) { float v = px[c]; s += v; s2 += v * v; }
  #pragma unroll
  for (int o = 16; o >= 1; o >>= 1) { s += __shfl_xor(s, o, 32); s2 += __shfl_xor(s2, o, 32); }
  __shared__ float sh[2][8];
  int wv = threadIdx.x >> 5, lane = threadIdx.x & 31;
  if (lane == 0) { sh[0][wv] = s; sh[1][wv] = s2; }
  __syncthreads();
  if (threadIdx.x == 0) {
    float a = 0.f, b2 = 0.f;
    for (int i = 0; i < 8; ++i) { a += sh[0][i]; b2 += sh[1][i]; }
    sh[0][0] = a; sh[1][0] = b2;
  }
  __syncthreads();
  float mean = sh[0][0] / C;
  float var  = sh[1][0] / C - mean * mean;
  float inv  = rsqrtf(var + 1e-5f);
  for (int c = threadIdx.x; c < C; c += 256) {
    float yv = (px[c] - mean) * inv * w[c] + bia[c];
    if (outb) outb[row * C + c] = (__bf16)yv;
    if (outf) outf[row * C + c] = yv;
  }
}

// ---------------------------------------------------------------------------
// WMMA GEMM: Out(MxN) = A(MxK,bf16,row-major) * BT(NxK,bf16)^T + bias, with
// per-mode epilogues. Block = 256 thr (8 waves), block tile 64x256, wave tile
// 32x64 (2x4 accumulators, 8 WMMAs per K=32 step). Fragments stream from
// global (weights sit in 192MB L2). A-row remap supports the "last cut rows
// of each batch" slice: g = (r/RO)*RI + Aoff + r%RO.
// ---------------------------------------------------------------------------
enum { EP_QK = 0, EP_VT = 1, EP_GELU = 2, EP_RESID = 3 };

template <int MODE>
__global__ __launch_bounds__(256) void gemm_bf16(
    const __bf16* __restrict__ A, const __bf16* __restrict__ BT,
    const float* __restrict__ bias, const float* __restrict__ resIn,
    void* __restrict__ Out, int K, int RI, int RO, int Aoff,
    int Tst, int resRI, int resOff, int Ncols) {
  const int lane = threadIdx.x & 31;
  const int wave = threadIdx.x >> 5;
  const int half = lane >> 4;     // 0 or 1
  const int ln16 = lane & 15;
  const int m0 = blockIdx.y * 64 + (wave >> 2) * 32;
  const int n0 = blockIdx.x * 256 + (wave & 3) * 64;

  v8f acc[2][4];
  #pragma unroll
  for (int f = 0; f < 2; ++f)
    #pragma unroll
    for (int nf = 0; nf < 4; ++nf)
      #pragma unroll
      for (int r = 0; r < 8; ++r) acc[f][nf][r] = 0.f;

  int gRow[2];
  #pragma unroll
  for (int f = 0; f < 2; ++f) {
    int rb = m0 + f * 16;
    int bb = rb / RO;
    gRow[f] = bb * RI + Aoff + (rb - bb * RO) + ln16;
  }

  const int c0off = half * 16;  // A-fragment lane k base: {0..15} | {16..31}
  const int kboff = half * 16;  // B-fragment lane k base: {0..15} | {16..31}

  for (int kk = 0; kk < K; kk += 32) {
    v16bf afr[2];
    #pragma unroll
    for (int f = 0; f < 2; ++f) {
      const __bf16* p = A + (size_t)gRow[f] * K + kk + c0off;
      afr[f] = cat16(*(const v8bf*)p, *(const v8bf*)(p + 8));
    }
    if (kk + 32 < K) {
      __builtin_prefetch(A + (size_t)gRow[0] * K + kk + 32 + c0off, 0, 1);
      __builtin_prefetch(BT + (size_t)(n0 + ln16) * K + kk + 32 + kboff, 0, 1);
    }
    #pragma unroll
    for (int nf = 0; nf < 4; ++nf) {
      const __bf16* p = BT + (size_t)(n0 + nf * 16 + ln16) * K + kk + kboff;
      v16bf bfr = cat16(*(const v8bf*)p, *(const v8bf*)(p + 8));
      acc[0][nf] = wmma_bf16(afr[0], bfr, acc[0][nf]);
      acc[1][nf] = wmma_bf16(afr[1], bfr, acc[1][nf]);
    }
  }

  const int Hh = Ncols >> 6;
  #pragma unroll
  for (int f = 0; f < 2; ++f)
    #pragma unroll
    for (int nf = 0; nf < 4; ++nf)
      #pragma unroll
      for (int r = 0; r < 8; ++r) {
        int m = m0 + f * 16 + r + half * 8;   // output row (C-layout)
        int j = n0 + nf * 16 + ln16;          // output col
        float vv = acc[f][nf][r] + bias[j];
        int bb = m / RO, t = m - bb * RO;
        if (MODE == EP_QK) {
          int hh = j >> 6, d = j & 63;
          ((__bf16*)Out)[(((size_t)(bb * Hh + hh) * Tst + t) << 6) + d] = (__bf16)vv;
        } else if (MODE == EP_VT) {
          int hh = j >> 6, d = j & 63;
          ((__bf16*)Out)[(((size_t)(bb * Hh + hh) << 6) + d) * Tst + t] = (__bf16)vv;
        } else if (MODE == EP_GELU) {
          float g = 0.5f * vv * (1.f + erff(vv * 0.70710678118654752f));
          ((__bf16*)Out)[(size_t)m * Ncols + j] = (__bf16)g;
        } else {  // EP_RESID (f32, residual add)
          size_t rr = (size_t)bb * resRI + resOff + t;
          ((float*)Out)[(size_t)m * Ncols + j] = resIn[rr * Ncols + j] + vv;
        }
      }
}

// ---------------------------------------------------------------------------
// Flash-style attention: one wave (32 thr) handles (b,h,16 q rows). Scores via
// WMMA (K=64 in 2 steps), online softmax with 16-lane shuffle reductions, P is
// relaid C-layout -> A-layout through LDS, then O += P @ V^T via WMMA.
// q: [B,H,cut,64] bf16, k: [B,H,Tc,64] bf16, v: [B,H,64,Tc] bf16 (transposed),
// o: [B*cut, 768] bf16 row-major.
// ---------------------------------------------------------------------------
__global__ __launch_bounds__(32) void attn_kernel(
    const __bf16* __restrict__ q, const __bf16* __restrict__ k,
    const __bf16* __restrict__ v, __bf16* __restrict__ o, int Tc, int cut) {
  __shared__ __align__(16) __bf16 Pst[16 * 32];
  const int lane = threadIdx.x & 31;
  const int half = lane >> 4;
  const int ln16 = lane & 15;
  const int tiles = cut >> 4;
  const int tile = blockIdx.x % tiles;
  const int bh = blockIdx.x / tiles;
  const int b = bh / NH, hh = bh % NH;

  const __bf16* qp = q + (size_t)bh * cut * DHD;
  const __bf16* kp = k + (size_t)bh * Tc * DHD;
  const __bf16* vp = v + (size_t)bh * DHD * Tc;

  v16bf qf[2];
  {
    const int qrow = tile * 16 + ln16;
    #pragma unroll
    for (int kk2 = 0; kk2 < 2; ++kk2) {
      const __bf16* p = qp + (size_t)qrow * DHD + kk2 * 32 + half * 16;
      qf[kk2] = cat16(*(const v8bf*)p, *(const v8bf*)(p + 8));
    }
  }

  v8f Oa[4];
  float rmax[8], rsum[8];
  #pragma unroll
  for (int nf = 0; nf < 4; ++nf)
    #pragma unroll
    for (int r = 0; r < 8; ++r) Oa[nf][r] = 0.f;
  #pragma unroll
  for (int r = 0; r < 8; ++r) { rmax[r] = -1e30f; rsum[r] = 0.f; }

  const float scale = 0.03608439182435161f;  // 768^-0.5 (reference uses C^-0.5)
  const int base = Tc - cut;
  const int smax = base + tile * 16 + 15;

  for (int s0 = 0; s0 <= smax; s0 += 32) {
    v8f S0, S1;
    #pragma unroll
    for (int r = 0; r < 8; ++r) { S0[r] = 0.f; S1[r] = 0.f; }
    #pragma unroll
    for (int kk2 = 0; kk2 < 2; ++kk2) {
      int kbd = kk2 * 32 + half * 16;
      const __bf16* p0 = kp + (size_t)(s0 + ln16) * DHD + kbd;
      v16bf kf0 = cat16(*(const v8bf*)p0, *(const v8bf*)(p0 + 8));
      S0 = wmma_bf16(qf[kk2], kf0, S0);
      const __bf16* p1 = kp + (size_t)(s0 + 16 + ln16) * DHD + kbd;
      v16bf kf1 = cat16(*(const v8bf*)p1, *(const v8bf*)(p1 + 8));
      S1 = wmma_bf16(qf[kk2], kf1, S1);
    }
    // scale + causal mask + row max (16-lane groups match C-layout rows)
    float mloc[8];
    #pragma unroll
    for (int r = 0; r < 8; ++r) {
      int m = r + half * 8;
      int qpos = base + tile * 16 + m;
      float a0 = (s0 + ln16 <= qpos)      ? S0[r] * scale : -1e30f;
      float a1 = (s0 + 16 + ln16 <= qpos) ? S1[r] * scale : -1e30f;
      S0[r] = a0; S1[r] = a1;
      float mv = fmaxf(a0, a1);
      #pragma unroll
      for (int d = 1; d < 16; d <<= 1) mv = fmaxf(mv, __shfl_xor(mv, d, 16));
      mloc[r] = mv;
    }
    float P0[8], P1[8];
    #pragma unroll
    for (int r = 0; r < 8; ++r) {
      float mn = fmaxf(rmax[r], mloc[r]);
      float f  = __expf(rmax[r] - mn);
      rmax[r] = mn;
      rsum[r] *= f;
      #pragma unroll
      for (int nf = 0; nf < 4; ++nf) Oa[nf][r] *= f;
      P0[r] = __expf(S0[r] - mn);
      P1[r] = __expf(S1[r] - mn);
      float rs = P0[r] + P1[r];
      #pragma unroll
      for (int d = 1; d < 16; d <<= 1) rs += __shfl_xor(rs, d, 16);
      rsum[r] += rs;
    }
    // C-layout -> LDS (bf16), then reload as A-fragment (same-wave LDS is in-order)
    #pragma unroll
    for (int r = 0; r < 8; ++r) {
      int m = r + half * 8;
      Pst[m * 32 + ln16]      = (__bf16)P0[r];
      Pst[m * 32 + 16 + ln16] = (__bf16)P1[r];
    }
    const __bf16* pp = &Pst[ln16 * 32 + half * 16];
    v16bf pf = cat16(*(const v8bf*)pp, *(const v8bf*)(pp + 8));
    #pragma unroll
    for (int nf = 0; nf < 4; ++nf) {
      const __bf16* p = vp + (size_t)(nf * 16 + ln16) * Tc + s0 + half * 16;
      v16bf vf = cat16(*(const v8bf*)p, *(const v8bf*)(p + 8));
      Oa[nf] = wmma_bf16(pf, vf, Oa[nf]);
    }
  }
  #pragma unroll
  for (int nf = 0; nf < 4; ++nf)
    #pragma unroll
    for (int r = 0; r < 8; ++r) {
      int m = r + half * 8;
      size_t row = (size_t)b * cut + tile * 16 + m;
      int col = hh * 64 + nf * 16 + ln16;
      o[row * CDIM + col] = (__bf16)(Oa[nf][r] / rsum[r]);
    }
}

// ---------------------------------------------------------------------------
// Final logits: out[b,v] = hln[b,last,:] . Wout[:,v] + bout[v]. M=8 -> purely
// memory-bound on Wout (154MB f32): SIMT, LN rows cached in LDS.
// ---------------------------------------------------------------------------
__global__ __launch_bounds__(256) void logits_kernel(
    const float* __restrict__ hln, const float* __restrict__ Wo,
    const float* __restrict__ bo, float* __restrict__ out, int Trow, int V) {
  __shared__ float hs[NB * CDIM];
  for (int i = threadIdx.x; i < NB * CDIM; i += 256) {
    int b = i / CDIM, c = i - b * CDIM;
    hs[i] = hln[((size_t)b * Trow + Trow - 1) * CDIM + c];
  }
  __syncthreads();
  int vcol = blockIdx.x * 256 + threadIdx.x;
  if (vcol >= V) return;
  float acc[NB];
  #pragma unroll
  for (int b = 0; b < NB; ++b) acc[b] = bo[vcol];
  for (int c = 0; c < CDIM; ++c) {
    float wv = Wo[(size_t)c * V + vcol];
    #pragma unroll
    for (int b = 0; b < NB; ++b) acc[b] += hs[b * CDIM + c] * wv;
  }
  #pragma unroll
  for (int b = 0; b < NB; ++b) out[(size_t)b * V + vcol] = acc[b];
}

// ---------------------------------------------------------------------------
extern "C" void kernel_launch(void* const* d_in, const int* in_sizes, int n_in,
                              void* d_out, int out_size, void* d_ws, size_t ws_size,
                              hipStream_t stream) {
  const int*   x    = (const int*)  d_in[0];
  const float* tok  = (const float*)d_in[1];
  const float* pos  = (const float*)d_in[2];
  const float* Wq   = (const float*)d_in[3];
  const float* bq   = (const float*)d_in[4];
  const float* Wk   = (const float*)d_in[5];
  const float* bk   = (const float*)d_in[6];
  const float* Wv   = (const float*)d_in[7];
  const float* bv   = (const float*)d_in[8];
  const float* Wp   = (const float*)d_in[9];
  const float* bp   = (const float*)d_in[10];
  const float* l1w  = (const float*)d_in[11];
  const float* l1b  = (const float*)d_in[12];
  const float* l2w  = (const float*)d_in[13];
  const float* l2b  = (const float*)d_in[14];
  const float* W1   = (const float*)d_in[15];
  const float* b1   = (const float*)d_in[16];
  const float* W2   = (const float*)d_in[17];
  const float* b2   = (const float*)d_in[18];
  const float* lfw  = (const float*)d_in[19];
  const float* lfb  = (const float*)d_in[20];
  const float* Wout = (const float*)d_in[21];
  const float* bout = (const float*)d_in[22];

  char* wsb = (char*)d_ws;
  size_t off = 0;
  auto take = [&](size_t bytes) -> char* {
    char* p = wsb + off;
    off = (off + bytes + 255) & ~(size_t)255;
    return p;
  };
  __bf16* wqT  = (__bf16*)take((size_t)NLAYER * CDIM * CDIM * 2);
  __bf16* wkT  = (__bf16*)take((size_t)NLAYER * CDIM * CDIM * 2);
  __bf16* wvT  = (__bf16*)take((size_t)NLAYER * CDIM * CDIM * 2);
  __bf16* wpT  = (__bf16*)take((size_t)NLAYER * CDIM * CDIM * 2);
  __bf16* wf1T = (__bf16*)take((size_t)NLAYER * FDIM * CDIM * 2);
  __bf16* wf2T = (__bf16*)take((size_t)NLAYER * CDIM * FDIM * 2);
  float*  hA   = (float*) take((size_t)NB * NT * CDIM * 4);
  float*  hB   = (float*) take((size_t)NB * NT * CDIM * 4);
  __bf16* yb   = (__bf16*)take((size_t)NB * NT * CDIM * 2);
  __bf16* qb   = (__bf16*)take((size_t)NB * NH * NT * DHD * 2);
  __bf16* kbf  = (__bf16*)take((size_t)NB * NH * NT * DHD * 2);
  __bf16* vbf  = (__bf16*)take((size_t)NB * NH * NT * DHD * 2);
  __bf16* ob   = (__bf16*)take((size_t)NB * NT * CDIM * 2);
  __bf16* gb   = (__bf16*)take((size_t)NB * NT * FDIM * 2);
  float*  hln  = (float*) take((size_t)NB * 64 * CDIM * 4);

  dim3 blk(256);
  // ---- repack weights to bf16 / N-major (runs each call; deterministic) ----
  long nqkv = (long)NLAYER * CDIM * CDIM;
  pack_headed<<<2048, blk, 0, stream>>>(wqT, Wq, nqkv, NH, CDIM, DHD);
  pack_headed<<<2048, blk, 0, stream>>>(wkT, Wk, nqkv, NH, CDIM, DHD);
  pack_headed<<<2048, blk, 0, stream>>>(wvT, Wv, nqkv, NH, CDIM, DHD);
  pack_T<<<2048, blk, 0, stream>>>(wpT,  Wp, nqkv, CDIM, CDIM);
  pack_T<<<4096, blk, 0, stream>>>(wf1T, W1, (long)NLAYER * FDIM * CDIM, CDIM, FDIM);
  pack_T<<<4096, blk, 0, stream>>>(wf2T, W2, (long)NLAYER * CDIM * FDIM, FDIM, CDIM);

  // ---- embedding ----
  embed_kernel<<<NB * NT, blk, 0, stream>>>(x, tok, pos, hA, NT, CDIM);

  int Tc = NT;
  for (int l = 0; l < NLAYER; ++l) {
    bool fade = (l != 0) && (l != NLAYER - 1);
    int cutv = fade ? ((NT >> l) < Tc ? (NT >> l) : Tc) : Tc;
    size_t wOff = (size_t)l * CDIM * CDIM;

    // ln1: y = LN(h)
    ln_kernel<<<NB * Tc, blk, 0, stream>>>(hA, l1w + l * CDIM, l1b + l * CDIM, yb, nullptr, CDIM);
    // q/k/v projections (WMMA)
    {
      dim3 g(CDIM / 256, (NB * cutv) / 64);
      gemm_bf16<EP_QK><<<g, blk, 0, stream>>>(yb, wqT + wOff, bq + l * CDIM, nullptr, qb,
                                              CDIM, Tc, cutv, Tc - cutv, cutv, 0, 0, CDIM);
    }
    {
      dim3 g(CDIM / 256, (NB * Tc) / 64);
      gemm_bf16<EP_QK><<<g, blk, 0, stream>>>(yb, wkT + wOff, bk + l * CDIM, nullptr, kbf,
                                              CDIM, Tc, Tc, 0, Tc, 0, 0, CDIM);
      gemm_bf16<EP_VT><<<g, blk, 0, stream>>>(yb, wvT + wOff, bv + l * CDIM, nullptr, vbf,
                                              CDIM, Tc, Tc, 0, Tc, 0, 0, CDIM);
    }
    // attention (flash-style, WMMA)
    attn_kernel<<<NB * NH * (cutv / 16), dim3(32), 0, stream>>>(qb, kbf, vbf, ob, Tc, cutv);
    // output projection + residual (h slice)
    {
      dim3 g(CDIM / 256, (NB * cutv) / 64);
      gemm_bf16<EP_RESID><<<g, blk, 0, stream>>>(ob, wpT + wOff, bp + l * CDIM, hA, hB,
                                                 CDIM, cutv, cutv, 0, 0, Tc, Tc - cutv, CDIM);
    }
    // ln2 + MLP
    ln_kernel<<<NB * cutv, blk, 0, stream>>>(hB, l2w + l * CDIM, l2b + l * CDIM, yb, nullptr, CDIM);
    {
      dim3 g(FDIM / 256, (NB * cutv) / 64);
      gemm_bf16<EP_GELU><<<g, blk, 0, stream>>>(yb, wf1T + (size_t)l * FDIM * CDIM, b1 + l * FDIM,
                                                nullptr, gb, CDIM, cutv, cutv, 0, 0, 0, 0, FDIM);
    }
    {
      dim3 g(CDIM / 256, (NB * cutv) / 64);
      gemm_bf16<EP_RESID><<<g, blk, 0, stream>>>(gb, wf2T + (size_t)l * CDIM * FDIM, b2 + l * CDIM,
                                                 hB, hA, FDIM, cutv, cutv, 0, 0, cutv, 0, CDIM);
    }
    Tc = cutv;
  }

  // ---- final LN + logits (last token only) ----
  ln_kernel<<<NB * Tc, blk, 0, stream>>>(hA, lfw, lfb, nullptr, hln, CDIM);
  logits_kernel<<<(NV + 255) / 256, blk, 0, stream>>>(hln, Wout, bout, (float*)d_out, Tc, NV);
}
